// ResidualGCNBlock_40535901339691
// MI455X (gfx1250) — matrix operand order repaired and checked by reading, not compiled, not run on hardware
//
#include <hip/hip_runtime.h>
#include <hip/hip_fp16.h>

// MI455X / gfx1250, wave32.
// GEMMs: v_wmma_f32_16x16x32_f16 on fragment-packed f16 operands
//        (2x global_load_b128 per fragment, 1 wmma per K-block of 32).
// Aggregation (dominant ~7GB irregular traffic): fp32 global atomics.
// All runtime divisions replaced by shifts / grid mapping.

typedef __attribute__((ext_vector_type(16))) _Float16 v16h;
typedef __attribute__((ext_vector_type(8)))  float    v8f;

#define WG 256

// ---------------- degree / norm ----------------
__global__ __launch_bounds__(WG) void k_deg_init(float* deg, int n) {
  int i = blockIdx.x * WG + threadIdx.x;
  if (i < n) deg[i] = 1.0f;                       // self-loop contributes 1
}

__global__ __launch_bounds__(WG) void k_deg_count(const int* __restrict__ dst,
                                                  float* deg, long long e) {
  long long i = (long long)blockIdx.x * WG + threadIdx.x;
  if (i < e) atomicAdd(&deg[dst[i]], 1.0f);
}

__global__ __launch_bounds__(WG) void k_dinv(float* deg, int n) {
  int i = blockIdx.x * WG + threadIdx.x;
  if (i < n) { float d = deg[i]; deg[i] = d > 0.f ? rsqrtf(d) : 0.f; }
}

// ---------------- fragment packing (CDNA5 ISA 7.12.2 layouts) ----------------
// A 16x32 f16 fragment: lane = g*16 + (row%16); VGPR v holds
//   K = (v<4 ? 2v : 16+2(v-4)) + 8g  and K+1.
// Packed: P[(( (row/16)<<kbBits | kb )*32 + lane)*16 + elem], elem = 2v+j.
__global__ __launch_bounds__(WG)
void k_pack_a(const float* __restrict__ A, _Float16* __restrict__ P,
              int M, int K, int kbBits, long long total) {
  long long gid = (long long)blockIdx.x * WG + threadIdx.x;
  if (gid >= total) return;
  int elem = (int)(gid & 15);
  int lane = (int)((gid >> 4) & 31);
  long long t = gid >> 9;                 // t = mt<<kbBits | kb
  int kb = (int)(t & ((1 << kbBits) - 1));
  int mt = (int)(t >> kbBits);
  int g = lane >> 4, r15 = lane & 15;
  int v = elem >> 1, j = elem & 1;
  int kin = ((v < 4) ? 2 * v : 16 + 2 * (v - 4)) + 8 * g + j;
  int row = mt * 16 + r15;
  int k = kb * 32 + kin;
  P[gid] = (row < M) ? (_Float16)A[(long long)row * K + k] : (_Float16)0.f;
}

// B 32x16 f16 fragment: lane = g*16 + (col%16); VGPR v holds K = (v<4 ? 2v : 16+2(v-4)) + 8g, +1.
__global__ __launch_bounds__(WG)
void k_pack_b(const float* __restrict__ B, _Float16* __restrict__ P,
              int K, int Nn, int kbBits, long long total) {
  long long gid = (long long)blockIdx.x * WG + threadIdx.x;
  if (gid >= total) return;
  int elem = (int)(gid & 15);
  int lane = (int)((gid >> 4) & 31);
  long long t = gid >> 9;                 // t = nt<<kbBits | kb
  int kb = (int)(t & ((1 << kbBits) - 1));
  int nt = (int)(t >> kbBits);
  int g = lane >> 4, r15 = lane & 15;
  int v = elem >> 1, j = elem & 1;
  int k = kb * 32 + ((v < 4) ? 2 * v : 16 + 2 * (v - 4)) + 8 * g + j;
  int col = nt * 16 + r15;
  P[gid] = (_Float16)B[(long long)k * Nn + col];
}

// ---------------- WMMA GEMM on packed operands ----------------
// 8 waves/block: 2 m-tiles x 4 n-tiles (32x64 of C).
__global__ __launch_bounds__(WG)
void k_gemm_packed(const _Float16* __restrict__ Ap, const _Float16* __restrict__ Bp,
                   const float* __restrict__ bias, float* __restrict__ C,
                   int M, int K, int Nn, int mtiles) {
  const int lane = threadIdx.x & 31;
  const int wave = threadIdx.x >> 5;
  const int KB = K >> 5;
  const int mt = blockIdx.x * 2 + (wave & 1);
  const int nt = blockIdx.y * 4 + (wave >> 1);
  if (mt >= mtiles) return;               // wave-uniform tail guard

  const v16h* __restrict__ Afrag = (const v16h*)Ap + (long long)mt * KB * 32 + lane;
  const v16h* __restrict__ Bfrag = (const v16h*)Bp + (long long)nt * KB * 32 + lane;

  v8f acc = {};
  for (int kb = 0; kb < KB; ++kb) {
    v16h a = Afrag[(long long)kb * 32];
    v16h b = Bfrag[(long long)kb * 32];
    acc = __builtin_amdgcn_wmma_f32_16x16x32_f16(false, a, false, b,
                                                 (short)0, acc, false, false);
  }

  const int g = lane >> 4, r15 = lane & 15;
  const int colB = nt * 16 + r15;
  const float bn = bias ? bias[colB] : 0.f;   // wave-uniform pointer test
  const int mbase = mt * 16 + g * 8;          // C/D: VGPR r -> row mbase+r
  float* Cb = C + (long long)mbase * Nn + colB;
  if (mt * 16 + 16 <= M) {                    // wave-uniform: tile fully interior
#pragma unroll
    for (int r = 0; r < 8; ++r) Cb[(long long)r * Nn] = acc[r] + bn;
  } else {
#pragma unroll
    for (int r = 0; r < 8; ++r)
      if (mbase + r < M) Cb[(long long)r * Nn] = acc[r] + bn;
  }
}

// ---------------- edge scatter: acc[dst] += feat[src] * dinv[src] -----------
// (nout/4) threads per edge, float4 gather, 4 f32 atomics; shift-based index.
__global__ __launch_bounds__(WG)
void k_scatter(const float* __restrict__ feat, const float* __restrict__ dinv,
               const int* __restrict__ src, const int* __restrict__ dst,
               float* __restrict__ acc, long long e, int nout, int perShift) {
  long long gid = (long long)blockIdx.x * WG + threadIdx.x;
  long long eidx = gid >> perShift;
  int q = (int)(gid & ((1 << perShift) - 1));
  if (eidx >= e) return;
  int s = src[eidx], d = dst[eidx];
  float sc = dinv[s];
  float4 v = ((const float4*)(feat + (long long)s * nout))[q];
  float* p = acc + (long long)d * nout + (long long)q * 4;
  atomicAdd(p + 0, v.x * sc);
  atomicAdd(p + 1, v.y * sc);
  atomicAdd(p + 2, v.z * sc);
  atomicAdd(p + 3, v.w * sc);
}

// ---------------- epilogues (row = blockIdx.x, channel = blockIdx.y*WG+tid) --
// h = relu((acc + xw*dinv)*dinv + b1); written DIRECTLY in packed-A layout.
__global__ __launch_bounds__(WG)
void k_finalize1(const float* __restrict__ acc, const float* __restrict__ xw,
                 const float* __restrict__ dinv, const float* __restrict__ bias,
                 _Float16* __restrict__ hp, int nout, int kbBits) {
  int i = blockIdx.x;
  int c = blockIdx.y * WG + threadIdx.x;
  long long gid = (long long)i * nout + c;
  float di = dinv[i];
  float t = (acc[gid] + xw[gid] * di) * di + bias[c];
  t = t > 0.f ? t : 0.f;
  // inverse of A-fragment map: c is the K index of the next GEMM
  int kb = c >> 5, kin = c & 31;
  int j = kin & 1, u = kin >> 1;
  int g = (u >> 2) & 1;
  int v = (u & 3) | ((u >> 3) << 2);
  int lane = g * 16 + (i & 15);
  int elem = 2 * v + j;
  long long pidx = ((((long long)(i >> 4) << kbBits) + kb) * 32 + lane) * 16 + elem;
  hp[pidx] = (_Float16)t;
}

// out = relu((out + hw2*dinv)*dinv + b2 + xwp)   (in-place on d_out)
__global__ __launch_bounds__(WG)
void k_finalize2(float* accout, const float* __restrict__ hw2,
                 const float* __restrict__ xwp, const float* __restrict__ dinv,
                 const float* __restrict__ bias, int nout) {
  int i = blockIdx.x;
  int c = blockIdx.y * WG + threadIdx.x;
  long long gid = (long long)i * nout + c;
  float di = dinv[i];
  float t = (accout[gid] + hw2[gid] * di) * di + bias[c] + xwp[gid];
  accout[gid] = t > 0.f ? t : 0.f;
}

static inline int ilog2i(int v) { int s = 0; while ((1 << s) < v) ++s; return s; }

// ---------------- launch ----------------
extern "C" void kernel_launch(void* const* d_in, const int* in_sizes, int n_in,
                              void* d_out, int out_size, void* d_ws, size_t ws_size,
                              hipStream_t stream) {
  const float* x  = (const float*)d_in[0];
  const int*   ei = (const int*)d_in[1];   // per harness contract: int indices
  const float* W1 = (const float*)d_in[2];
  const float* b1 = (const float*)d_in[3];
  const float* W2 = (const float*)d_in[4];
  const float* b2 = (const float*)d_in[5];
  const float* Wp = (const float*)d_in[6];
  const float* bp = (const float*)d_in[7];
  float* out = (float*)d_out;

  const int OUTC = in_sizes[3];              // 256
  const int INC  = in_sizes[2] / OUTC;       // 128
  const int N    = in_sizes[0] / INC;        // 100000
  const long long E = in_sizes[1] / 2;       // 1.6M
  const int* src = ei;
  const int* dst = ei + E;

  const int kb1 = ilog2i(INC >> 5);          // K-blocks bits for K=IN  (2)
  const int kb2 = ilog2i(OUTC >> 5);         // K-blocks bits for K=OUT (3)
  const int perShift = ilog2i(OUTC >> 2);    // scatter: threads-per-edge bits (6)

  const int mtiles = (N + 15) / 16;          // 6250 (exact for N=100000)
  const long long apTotal = (long long)mtiles * (INC  >> 5) * 512;
  const long long hpTotal = (long long)mtiles * (OUTC >> 5) * 512;
  const long long b1Total = (long long)(OUTC / 16) * (INC  >> 5) * 512;
  const long long b2Total = (long long)(OUTC / 16) * (OUTC >> 5) * 512;

  // ---- workspace carve (~385 MB) ----
  auto al = [](size_t v) { return (v + 255) & ~(size_t)255; };
  char* w = (char*)d_ws;
  float*    dinv = (float*)w;      w += al((size_t)N * 4);
  _Float16* xp   = (_Float16*)w;   w += al((size_t)apTotal * 2);
  _Float16* hp   = (_Float16*)w;   w += al((size_t)hpTotal * 2);
  float*    xw1  = (float*)w;      w += al((size_t)N * OUTC * 4);
  float*    xwp  = (float*)w;      w += al((size_t)N * OUTC * 4);
  float*    acc1 = (float*)w;      w += al((size_t)N * OUTC * 4);
  _Float16* W1p  = (_Float16*)w;   w += al((size_t)b1Total * 2);
  _Float16* W2p  = (_Float16*)w;   w += al((size_t)b2Total * 2);
  _Float16* Wpp  = (_Float16*)w;   w += al((size_t)b1Total * 2);
  float* hw2 = xw1;                // xw1 dead after k_finalize1; reuse

  dim3 blk(WG);
  auto cdiv = [](long long a, long long b) { return (unsigned)((a + b - 1) / b); };

  // normalization
  k_deg_init <<<cdiv(N, WG), blk, 0, stream>>>(dinv, N);
  k_deg_count<<<cdiv(E, WG), blk, 0, stream>>>(dst, dinv, E);
  k_dinv     <<<cdiv(N, WG), blk, 0, stream>>>(dinv, N);

  // pack operands into WMMA fragment layout (also f32 -> f16)
  k_pack_a<<<cdiv(apTotal, WG), blk, 0, stream>>>(x,  xp,  N, INC, kb1, apTotal);
  k_pack_b<<<cdiv(b1Total, WG), blk, 0, stream>>>(W1, W1p, INC,  OUTC, kb1, b1Total);
  k_pack_b<<<cdiv(b2Total, WG), blk, 0, stream>>>(W2, W2p, OUTC, OUTC, kb2, b2Total);
  k_pack_b<<<cdiv(b1Total, WG), blk, 0, stream>>>(Wp, Wpp, INC,  OUTC, kb1, b1Total);

  // GEMMs: xw1 = x@W1 ; xwp = x@Wp + bp
  dim3 g1(cdiv(mtiles, 2), OUTC / 64);
  k_gemm_packed<<<g1, blk, 0, stream>>>(xp, W1p, nullptr, xw1, N, INC, OUTC, mtiles);
  k_gemm_packed<<<g1, blk, 0, stream>>>(xp, Wpp, bp,      xwp, N, INC, OUTC, mtiles);

  // layer 1 aggregation + fused epilogue -> hp (f16, packed-A layout)
  hipMemsetAsync(acc1, 0, (size_t)N * OUTC * 4, stream);
  if (mtiles * 16 != N)                         // pad rows would be poison
    hipMemsetAsync(hp, 0, (size_t)hpTotal * 2, stream);
  k_scatter<<<cdiv(E << perShift, WG), blk, 0, stream>>>(xw1, dinv, src, dst, acc1,
                                                         E, OUTC, perShift);
  dim3 gf(N, OUTC / WG);
  k_finalize1<<<gf, blk, 0, stream>>>(acc1, xw1, dinv, b1, hp, OUTC, kb2);

  // layer 2 GEMM: hw2 = h@W2
  k_gemm_packed<<<g1, blk, 0, stream>>>(hp, W2p, nullptr, hw2, N, OUTC, OUTC, mtiles);

  // layer 2 aggregation directly into d_out + fused residual epilogue
  hipMemsetAsync(out, 0, (size_t)N * OUTC * 4, stream);
  k_scatter<<<cdiv(E << perShift, WG), blk, 0, stream>>>(hw2, dinv, src, dst, out,
                                                         E, OUTC, perShift);
  k_finalize2<<<gf, blk, 0, stream>>>(out, hw2, xwp, dinv, b2, OUTC);
}
